// Tracker_14310831030693
// MI455X (gfx1250) — matrix operand order repaired and checked by reading, not compiled, not run on hardware
//
#include <hip/hip_runtime.h>
#include <hip/hip_bf16.h>
#include <math.h>

// ---- problem dims ----
#define KP    32      // particles
#define BB    32      // batch
#define TT    64      // seq len
#define HH    256     // hidden
#define EE    256     // cell input (128 obs + 128 act)
#define G5    1280    // 5*H gates
#define KB    1024    // KP*BB
#define KD    256     // GEMM reduction dim (== HH == EE)
#define OBSN  64
#define WIN   50
#define ALPHA 0.5f

typedef __attribute__((ext_vector_type(16))) _Float16 v16h;
typedef __attribute__((ext_vector_type(8)))  _Float16 v8h;
typedef __attribute__((ext_vector_type(8)))  float    v8f;

// ------------------ small device helpers ------------------
__device__ __forceinline__ float sigf(float x)  { return 1.0f / (1.0f + __expf(-x)); }
__device__ __forceinline__ float softplusf(float x) {
    return (x > 20.0f) ? x : log1pf(__expf(x));
}
__device__ __forceinline__ unsigned pcg_hash(unsigned v) {
    v = v * 747796405u + 2891336453u;
    unsigned w = ((v >> ((v >> 28) + 4u)) ^ v) * 277803737u;
    return (w >> 22) ^ w;
}
__device__ __forceinline__ float u01(unsigned h) {   // (0,1)
    return ((float)h + 0.5f) * 2.3283064365386963e-10f;
}
__device__ __forceinline__ float gauss(unsigned seed) {
    float u1 = u01(pcg_hash(seed));
    float u2 = u01(pcg_hash(seed ^ 0x9E3779B9u));
    return sqrtf(-2.0f * __logf(u1)) * __cosf(6.283185307179586f * u2);
}
__device__ __forceinline__ float blockReduceSum256(float v, float* sbuf) {
    int tid = threadIdx.x;
    sbuf[tid] = v; __syncthreads();
    #pragma unroll
    for (int s = 128; s > 0; s >>= 1) {
        if (tid < s) sbuf[tid] += sbuf[tid + s];
        __syncthreads();
    }
    float r = sbuf[0]; __syncthreads();
    return r;
}

// ------------------ setup kernels ------------------
// Wt[k*G5+n] = (f16) W[n*HH + k]  (transpose + fp16 convert)
__global__ void transpose_f16_kernel(const float* __restrict__ W, _Float16* __restrict__ Wt) {
    int i = blockIdx.x * blockDim.x + threadIdx.x;
    if (i >= HH * G5) return;
    int k = i / G5, n = i - k * G5;
    Wt[i] = (_Float16)W[n * HH + k];
}

__global__ void bias_sum_kernel(const float* __restrict__ b_ih, const float* __restrict__ b_hh,
                                float* __restrict__ bsum) {
    int i = blockIdx.x * blockDim.x + threadIdx.x;
    if (i < G5) bsum[i] = b_ih[i] + b_hh[i];
}

__global__ void init_state_kernel(const float* __restrict__ h0, const float* __restrict__ c0,
                                  _Float16* __restrict__ h16, float* __restrict__ c,
                                  float* __restrict__ p) {
    int i = blockIdx.x * blockDim.x + threadIdx.x;
    if (i < KB * HH) {
        h16[i] = (_Float16)h0[i];
        c[i]   = c0[i];
    }
    if (i < KB) p[i] = __logf(1.0f / (float)KP);
}

// one block per (b,t); 256 threads -> 256 embedding channels.
__global__ void embed_kernel(const float* __restrict__ obs_in, const float* __restrict__ prev_window,
                             const float* __restrict__ W_obs, const float* __restrict__ b_obs,
                             const float* __restrict__ W_act, const float* __restrict__ b_act,
                             const float* __restrict__ W_fcobs, const float* __restrict__ b_fcobs,
                             _Float16* __restrict__ x16, float* __restrict__ xdot) {
    __shared__ float sbuf[256];
    int bt = blockIdx.x;               // b*T + t
    int b = bt / TT, t = bt - b * TT;
    int e = threadIdx.x;
    float val;
    if (e < 128) {
        const float* o = obs_in + (size_t)bt * OBSN;
        const float* w = W_obs + (size_t)e * OBSN;
        float acc = b_obs[e];
        #pragma unroll 4
        for (int i = 0; i < OBSN; ++i) acc += o[i] * w[i];
        val = fmaxf(acc, 0.0f);
    } else {
        int ea = e - 128;
        const float* a = prev_window + (size_t)bt * WIN;
        const float* w = W_act + (size_t)ea * WIN;
        float acc = b_act[ea];
        for (int i = 0; i < WIN; ++i) acc += a[i] * w[i];
        val = fmaxf(acc, 0.0f);
    }
    size_t row = (size_t)t * BB + b;
    x16[row * EE + e] = (_Float16)val;
    float red = blockReduceSum256(val * W_fcobs[HH + e], sbuf);
    if (e == 0) xdot[row] = red + b_fcobs[0];
}

// ------------------ WMMA GEMM: C[M,N] = A[M,256] * Bw[256,N] (+ bias[n]) ------------------
// grid: (N/256, M/16), block = 128 threads (4 waves).
// Block stages its 16x256 f16 A tile in LDS once; each wave computes 4 adjacent
// 16x16 N-tiles (4 independent WMMA accumulator chains share one A fragment).
__global__ void gemm_f16_wmma(const _Float16* __restrict__ A, const _Float16* __restrict__ Bw,
                              const float* __restrict__ bias, float* __restrict__ C,
                              int N) {
    __shared__ _Float16 sA[16 * KD];   // 8 KB
    int tm   = blockIdx.y;
    int wv   = threadIdx.x >> 5;       // wave in block: 0..3
    int lane = threadIdx.x & 31;
    // ---- cooperative A tile stage: 8192 B = 512 x 16B chunks, 128 threads x 4 ----
    {
        const float4* src = (const float4*)(A + (size_t)(tm << 4) * KD);
        float4* dst = (float4*)sA;
        #pragma unroll
        for (int i = 0; i < 4; ++i)
            dst[threadIdx.x + i * 128] = src[threadIdx.x + i * 128];
    }
    __syncthreads();
    int half = lane >> 4;
    int l15  = lane & 15;
    int n4   = blockIdx.x * 4 + wv;    // group of 4 n-tiles; base col = n4*64
    int col0 = n4 << 6;
    const _Float16* arow = sA + (size_t)l15 * KD;
    v8f acc0 = {}, acc1 = {}, acc2 = {}, acc3 = {};
    for (int k0 = 0; k0 < KD; k0 += 32) {
        // A fragment (16x32): lane-half selects K-phase; two contiguous 8-half runs (LDS)
        v8h alo = *(const v8h*)(arow + k0 + half * 8);
        v8h ahi = *(const v8h*)(arow + k0 + 16 + half * 8);
        v16h a;
        #pragma unroll
        for (int i = 0; i < 8; ++i) { a[i] = alo[i]; a[8 + i] = ahi[i]; }
        // B fragments (32x16 each): lane = N (col), elements = K in A's order; 4 adjacent tiles
        const _Float16* brow = Bw + (size_t)k0 * N + col0 + l15;
        if (k0 + 32 < KD)
            __builtin_prefetch((const void*)(brow + (size_t)32 * N), 0, 1);
        v16h b0t, b1t, b2t, b3t;
        {
            #pragma unroll
            for (int i = 0; i < 8; ++i) {
                size_t ka = (size_t)(half * 8 + i) * N, kc = (size_t)(16 + half * 8 + i) * N;
                b0t[i] = brow[ka];      b0t[8 + i] = brow[kc];
                b1t[i] = brow[ka + 16]; b1t[8 + i] = brow[kc + 16];
                b2t[i] = brow[ka + 32]; b2t[8 + i] = brow[kc + 32];
                b3t[i] = brow[ka + 48]; b3t[8 + i] = brow[kc + 48];
            }
        }
        acc0 = __builtin_amdgcn_wmma_f32_16x16x32_f16(false, a, false, b0t, (short)0, acc0, false, false);
        acc1 = __builtin_amdgcn_wmma_f32_16x16x32_f16(false, a, false, b1t, (short)0, acc1, false, false);
        acc2 = __builtin_amdgcn_wmma_f32_16x16x32_f16(false, a, false, b2t, (short)0, acc2, false, false);
        acc3 = __builtin_amdgcn_wmma_f32_16x16x32_f16(false, a, false, b3t, (short)0, acc3, false, false);
    }
    // D layout: element r -> row tm*16 + r + 8*half, col tile_base + (lane&15)
    size_t rowbase = (size_t)(tm << 4) + half * 8;
    #pragma unroll
    for (int j = 0; j < 4; ++j) {
        const v8f& acc = (j == 0) ? acc0 : (j == 1) ? acc1 : (j == 2) ? acc2 : acc3;
        int col = col0 + (j << 4) + l15;
        float bv = bias ? bias[col] : 0.0f;
        #pragma unroll
        for (int r = 0; r < 8; ++r)
            C[(rowbase + r) * N + col] = acc[r] + bv;
    }
}

// ------------------ per-step PF-LSTM cell ------------------
__global__ void pf_cell_kernel(int t,
                               const float* __restrict__ xg,  // [T*B, 1280] (biases folded)
                               const float* __restrict__ hg,  // [KB, 1280]
                               const float* __restrict__ c,   // [KB, H]
                               const float* __restrict__ xdot,// [T*B]
                               const float* __restrict__ W_fcobs,
                               float* __restrict__ h1buf, float* __restrict__ c1buf,
                               float* __restrict__ logpdf) {
    __shared__ float sbuf[256];
    int kb = blockIdx.x;
    int h  = threadIdx.x;
    int b  = kb & (BB - 1);
    const float* xr = xg + ((size_t)t * BB + b) * G5;
    const float* hr = hg + (size_t)kb * G5;
    float ii = xr[h]            + hr[h];
    float ff = xr[HH + h]       + hr[HH + h];
    float gg = xr[2 * HH + h]   + hr[2 * HH + h];
    float oo = xr[3 * HH + h]   + hr[3 * HH + h];
    float vv = xr[4 * HH + h]   + hr[4 * HH + h];
    float c1 = sigf(ff) * c[(size_t)kb * HH + h] + sigf(ii) * tanhf(gg);
    unsigned seed = ((unsigned)(t * KB + kb) * (unsigned)HH + (unsigned)h) * 2654435761u + 17u;
    c1 += gauss(seed) * softplusf(vv);
    float h1 = sigf(oo) * tanhf(c1);
    h1buf[(size_t)kb * HH + h] = h1;
    c1buf[(size_t)kb * HH + h] = c1;
    float red = blockReduceSum256(h1 * W_fcobs[h], sbuf);
    if (h == 0) logpdf[kb] = red + xdot[(size_t)t * BB + b];
}

// ------------------ per-step soft resampling ------------------
__global__ void pf_resample_kernel(int t,
                                   const float* __restrict__ logpdf,
                                   float* __restrict__ p,        // [KB] in/out log-weights
                                   int*   __restrict__ idxflat,  // [KB] gather indices
                                   float* __restrict__ pbuf) {   // [T, KB]
    __shared__ float sp1[KP], cdf[KP], red[KP];
    int b = blockIdx.x, k = threadIdx.x;
    int kb = k * BB + b;
    float lp = logpdf[kb] + p[kb];
    // logsumexp over particles
    red[k] = lp; __syncthreads();
    float m = red[0];
    for (int i = 1; i < KP; ++i) m = fmaxf(m, red[i]);
    red[k] = __expf(lp - m); __syncthreads();
    float s = 0.0f;
    for (int i = 0; i < KP; ++i) s += red[i];
    float p1 = lp - (m + __logf(s));
    sp1[k] = p1;
    float rp = ALPHA * __expf(p1) + (1.0f - ALPHA) / (float)KP;
    cdf[k] = rp; __syncthreads();
    // inclusive prefix sum (K=32, tiny)
    float acc = 0.0f;
    for (int i = 0; i <= k; ++i) acc += cdf[i];
    red[k] = acc; __syncthreads();
    float total = red[KP - 1];
    // draw category for output slot k
    unsigned seed = 0xC0FFEEu ^ ((unsigned)(t * BB + b) * 1023u + (unsigned)k * 7919u);
    float u = u01(pcg_hash(seed)) * total;
    int idx = KP - 1;
    for (int i = 0; i < KP; ++i) { if (u < red[i]) { idx = i; break; } }
    __syncthreads();
    // importance-weight correction, renormalize in log space
    float pe  = __expf(sp1[idx]);
    float lpn = __logf(pe / (ALPHA * pe + (1.0f - ALPHA) / (float)KP));
    red[k] = lpn; __syncthreads();
    float m2 = red[0];
    for (int i = 1; i < KP; ++i) m2 = fmaxf(m2, red[i]);
    float s2 = 0.0f;
    for (int i = 0; i < KP; ++i) s2 += __expf(red[i] - m2);
    float pn = lpn - (m2 + __logf(s2));
    p[kb] = pn;
    pbuf[(size_t)t * KB + kb] = pn;
    idxflat[kb] = idx * BB + b;
}

// ------------------ per-step gather + output-head dot ------------------
__global__ void pf_gather_kernel(int t,
                                 const int* __restrict__ idxflat,
                                 const float* __restrict__ h1buf, const float* __restrict__ c1buf,
                                 const float* __restrict__ W_lab,
                                 _Float16* __restrict__ h16, float* __restrict__ c,
                                 float* __restrict__ dbuf) {   // [T, KB] h.W_lab
    __shared__ float sbuf[256];
    int kb = blockIdx.x, h = threadIdx.x;
    int src = idxflat[kb];
    float hv = h1buf[(size_t)src * HH + h];
    float cv = c1buf[(size_t)src * HH + h];
    h16[(size_t)kb * HH + h] = (_Float16)hv;
    c[(size_t)kb * HH + h]   = cv;
    float red = blockReduceSum256(hv * W_lab[h], sbuf);
    if (h == 0) dbuf[(size_t)t * KB + kb] = red;
}

// ------------------ final outputs ------------------
__global__ void finalize_kernel(const float* __restrict__ dbuf, const float* __restrict__ pbuf,
                                const float* __restrict__ b_lab, float* __restrict__ out) {
    __shared__ float red[KP];
    int tb = blockIdx.x;
    int t = tb / BB, b = tb - t * BB;
    int k = threadIdx.x;
    int kb = k * BB + b;
    float dd = dbuf[(size_t)t * KB + kb];
    float w  = __expf(pbuf[(size_t)t * KB + kb]);
    float bl = b_lab[0];
    out[TT * BB + (size_t)t * KB + kb] = sigf(dd + bl);   // pf_out [T,KB,1]
    red[k] = w * dd; __syncthreads();
    if (k == 0) {
        float s = 0.0f;
        for (int i = 0; i < KP; ++i) s += red[i];
        out[(size_t)t * BB + b] = sigf(s + bl);            // y_out [T,B,1]
    }
}

// ------------------ host launch ------------------
extern "C" void kernel_launch(void* const* d_in, const int* in_sizes, int n_in,
                              void* d_out, int out_size, void* d_ws, size_t ws_size,
                              hipStream_t stream) {
    const float* obs_in      = (const float*)d_in[0];
    const float* prev_window = (const float*)d_in[1];
    const float* W_obs   = (const float*)d_in[2];
    const float* b_obs   = (const float*)d_in[3];
    const float* W_act   = (const float*)d_in[4];
    const float* b_act   = (const float*)d_in[5];
    const float* W_ih    = (const float*)d_in[6];
    const float* b_ih    = (const float*)d_in[7];
    const float* W_hh    = (const float*)d_in[8];
    const float* b_hh    = (const float*)d_in[9];
    const float* W_fcobs = (const float*)d_in[10];
    const float* b_fcobs = (const float*)d_in[11];
    const float* W_lab   = (const float*)d_in[12];
    const float* b_lab   = (const float*)d_in[13];
    const float* h0      = (const float*)d_in[14];
    const float* c0      = (const float*)d_in[15];
    float* out = (float*)d_out;

    char* ws = (char*)d_ws;
    size_t off = 0;
    auto carve = [&](size_t bytes) -> void* {
        off = (off + 255) & ~(size_t)255;
        void* p = ws + off;
        off += bytes;
        return p;
    };
    _Float16* x16   = (_Float16*)carve((size_t)TT * BB * EE * 2);
    float*    xdot  = (float*)   carve((size_t)TT * BB * 4);
    _Float16* WihT  = (_Float16*)carve((size_t)HH * G5 * 2);
    _Float16* WhhT  = (_Float16*)carve((size_t)HH * G5 * 2);
    float*    bsum  = (float*)   carve((size_t)G5 * 4);
    float*    xg    = (float*)   carve((size_t)TT * BB * G5 * 4);
    _Float16* h16   = (_Float16*)carve((size_t)KB * HH * 2);
    float*    cbuf  = (float*)   carve((size_t)KB * HH * 4);
    float*    pw    = (float*)   carve((size_t)KB * 4);
    float*    hg    = (float*)   carve((size_t)KB * G5 * 4);
    float*    h1b   = (float*)   carve((size_t)KB * HH * 4);
    float*    c1b   = (float*)   carve((size_t)KB * HH * 4);
    float*    lpdf  = (float*)   carve((size_t)KB * 4);
    int*      idxf  = (int*)     carve((size_t)KB * 4);
    float*    pbuf  = (float*)   carve((size_t)TT * KB * 4);
    float*    dbuf  = (float*)   carve((size_t)TT * KB * 4);
    (void)ws_size; (void)in_sizes; (void)n_in; (void)out_size;

    // setup: transposed f16 weights, fused bias, state init, input embeddings
    transpose_f16_kernel<<<(HH * G5 + 255) / 256, 256, 0, stream>>>(W_ih, WihT);
    transpose_f16_kernel<<<(HH * G5 + 255) / 256, 256, 0, stream>>>(W_hh, WhhT);
    bias_sum_kernel<<<(G5 + 255) / 256, 256, 0, stream>>>(b_ih, b_hh, bsum);
    init_state_kernel<<<(KB * HH + 255) / 256, 256, 0, stream>>>(h0, c0, h16, cbuf, pw);
    embed_kernel<<<BB * TT, 256, 0, stream>>>(obs_in, prev_window, W_obs, b_obs,
                                              W_act, b_act, W_fcobs, b_fcobs, x16, xdot);
    // xg = x @ W_ih.T + (b_ih + b_hh)   [T*B, 1280]
    gemm_f16_wmma<<<dim3(G5 / 256, TT * BB / 16), 128, 0, stream>>>(x16, WihT, bsum, xg, G5);
    // sequential PF-LSTM scan
    for (int t = 0; t < TT; ++t) {
        gemm_f16_wmma<<<dim3(G5 / 256, KB / 16), 128, 0, stream>>>(h16, WhhT, nullptr, hg, G5);
        pf_cell_kernel<<<KB, 256, 0, stream>>>(t, xg, hg, cbuf, xdot, W_fcobs, h1b, c1b, lpdf);
        pf_resample_kernel<<<BB, KP, 0, stream>>>(t, lpdf, pw, idxf, pbuf);
        pf_gather_kernel<<<KB, 256, 0, stream>>>(t, idxf, h1b, c1b, W_lab, h16, cbuf, dbuf);
    }
    // heads: y_out [T,B,1] then pf_out [T,KB,1]
    finalize_kernel<<<TT * BB, KP, 0, stream>>>(dbuf, pbuf, b_lab, out);
}
